// HarmonicAwaredAttention_63161789055302
// MI455X (gfx1250) — matrix operand order repaired and checked by reading, not compiled, not run on hardware
//
#include <hip/hip_runtime.h>
#include <hip/hip_bf16.h>

typedef __attribute__((ext_vector_type(16))) __bf16 bf16x16;
typedef __attribute__((ext_vector_type(8)))  __bf16 bf16x8;
typedef __attribute__((ext_vector_type(8)))  float  f32x8;

// ---- geometry ----
#define FB    128     // freq bins (seq len)
#define CH    128     // channels
#define NHEAD 2
#define DHEAD 64
#define NLAYER 4
#define TT    512
#define HS    136     // bf16 row stride: 272B rows -> 16B aligned, bank-conflict-free columns

// LDS byte offsets (dynamic shared). 7 x [128][136] bf16 buffers = 243,712 B (< 320KB WGP LDS)
// Weight staging aliases by liveness:
//   WqL -> Os slot, WkL -> P0 slot, WvL -> P1 slot (dead during QKV phase)
//   WoL -> Qs slot (dead after S-GEMM)
#define OFF_H   0
#define OFF_Q   34816
#define OFF_K   69632
#define OFF_VT  104448
#define OFF_O   139264
#define OFF_P   174080          // two head buffers: OFF_P + head*34816
#define SMEM_BYTES 243712

// Harmonic mask: |r-c| in {0,48,76,96,111,124} (offsets >=128 never occur for F=128)
#define MASK_LO 0x0001000000000001ULL
#define MASK_HI 0x1000800100001000ULL

// ---------- WMMA fragment helpers (CDNA5 bf16 16x16x32 layouts) ----------
// A 16x32 (row-major source): lanes 0-15 -> M=0..15, K chunks {0..7,16..23};
// lanes 16-31 -> K chunks {8..15,24..31}. Two 16B loads.
__device__ __forceinline__ bf16x16 frag_a(const __bf16* p, int stride, int m0, int k0, int lane) {
    int row = m0 + (lane & 15);
    int hb  = (lane & 16) ? 8 : 0;
    const __bf16* base = p + row * stride + k0 + hb;
    bf16x8 c0 = *(const bf16x8*)(base);
    bf16x8 c1 = *(const bf16x8*)(base + 16);
    bf16x16 r;
#pragma unroll
    for (int i = 0; i < 8; ++i) { r[i] = c0[i]; r[i + 8] = c1[i]; }
    return r;
}

// B 32x16 from column-major storage BT[n][k] (n-major, k contiguous):
// lanes 0-15 -> N=0..15, K chunks {0..7,16..23}; lanes 16-31 -> K chunks {8..15,24..31}. Two 16B loads.
__device__ __forceinline__ bf16x16 frag_b(const __bf16* p, int stride, int n0, int k0, int lane) {
    int col = n0 + (lane & 15);
    int hb  = (lane & 16) ? 8 : 0;
    const __bf16* base = p + col * stride + k0 + hb;
    bf16x8 c0 = *(const bf16x8*)(base);
    bf16x8 c1 = *(const bf16x8*)(base + 16);
    bf16x16 r;
#pragma unroll
    for (int i = 0; i < 8; ++i) { r[i] = c0[i]; r[i + 8] = c1[i]; }
    return r;
}

__device__ __forceinline__ f32x8 wmma_bf16(bf16x16 a, bf16x16 b, f32x8 c) {
    return __builtin_amdgcn_wmma_f32_16x16x32_bf16(false, a, false, b, (short)0, c, false, false);
}

// ---------- async global->LDS weight staging (CDNA5 TDM-lite path, ASYNCcnt) ----------
// Copies one 128x128 bf16 matrix (32KB, row-contiguous) into LDS with 272B padded rows.
// 512 threads x 4 rounds x 16B per lane.
__device__ __forceinline__ void stage_weight_async(const __bf16* gsrc, unsigned lds_base, int tid) {
#pragma unroll
    for (int rnd = 0; rnd < 4; ++rnd) {
        int g      = tid + (rnd << 9);             // 16B granule id, 0..2047
        int row    = g >> 4;                       // source row (256B = 16 granules)
        int within = (g & 15) << 4;
        unsigned lds = lds_base + (unsigned)(row * 272 + within);
        const char* gp = (const char*)gsrc + ((size_t)g << 4);
        asm volatile("global_load_async_to_lds_b128 %0, %1, off"
                     :: "v"(lds), "v"(gp) : "memory");
    }
}

__device__ __forceinline__ void wait_async_all() {
    asm volatile("s_wait_asynccnt 0" ::: "memory");
}

// ---------- weight prep: f32 [K=128][N=128] -> bf16 transposed [N=128][K=128] ----------
__global__ __launch_bounds__(256) void prep_weights(const float* __restrict__ Wq,
                                                    const float* __restrict__ Wk,
                                                    const float* __restrict__ Wv,
                                                    const float* __restrict__ Wo,
                                                    __bf16* __restrict__ wT) {
    int m  = blockIdx.x;          // 0..15 = l*4 + {q,k,v,o}
    int l  = m >> 2, ty = m & 3;
    const float* src = (ty == 0 ? Wq : ty == 1 ? Wk : ty == 2 ? Wv : Wo) + l * (CH * CH);
    __bf16* dst = wT + ((size_t)m << 14);
    for (int idx = threadIdx.x; idx < CH * CH; idx += 256) {
        int n = idx & 127, k = idx >> 7;
        dst[n * 128 + k] = (__bf16)src[k * 128 + n];   // coalesced read along n
    }
}

// ---------- fully fused 4-layer harmonic attention, one workgroup per (b,t) ----------
// 512 threads = 16 waves = 8 row-bands x 2 roles (col-half for GEMMs / head for attention)
__global__ __launch_bounds__(512) void ha_attn(const float* __restrict__ x,
                                               const __bf16* __restrict__ wT,
                                               const float* __restrict__ bq,
                                               const float* __restrict__ bk,
                                               const float* __restrict__ bv,
                                               const float* __restrict__ bo,
                                               float* __restrict__ out) {
    extern __shared__ __align__(16) char smem[];
    __bf16* Hs = (__bf16*)(smem + OFF_H);    // hidden  [128][HS]
    __bf16* Qs = (__bf16*)(smem + OFF_Q);    // pre-scaled by 1/sqrt(DH)
    __bf16* Ks = (__bf16*)(smem + OFF_K);    // row-major K == column-major B of Q.K^T
    __bf16* Vt = (__bf16*)(smem + OFF_VT);   // V transposed: Vt[d][f]
    __bf16* Os = (__bf16*)(smem + OFF_O);
    // staged weights (aliases, see liveness note at top)
    __bf16* WqL = (__bf16*)(smem + OFF_O);
    __bf16* WkL = (__bf16*)(smem + OFF_P);
    __bf16* WvL = (__bf16*)(smem + OFF_P + 34816);
    __bf16* WoL = (__bf16*)(smem + OFF_Q);

    const int n    = blockIdx.x;
    const int b    = n >> 9;                 // /512
    const int t    = n & 511;
    const int tid  = threadIdx.x;
    const int lane = tid & 31;
    const int W    = tid >> 5;               // wave 0..15
    const int w    = W & 7;                  // row band
    const int s    = W >> 3;                 // role / head (0,1)
    const int m0   = w << 4;
    const int ll   = lane & 15;
    const int hi8  = (lane & 16) ? 8 : 0;

    __bf16* Pss = (__bf16*)(smem + OFF_P + s * 34816);   // per-head probs

    // layer 0 Wq/Wk/Wv copies overlap the input staging below
    stage_weight_async(wT + 0 * 16384, (unsigned)OFF_O, tid);
    stage_weight_async(wT + 1 * 16384, (unsigned)OFF_P, tid);
    stage_weight_async(wT + 2 * 16384, (unsigned)(OFF_P + 34816), tid);

    // ---- stage input + fixed sinusoidal positional embedding ----
    const float k10k = 9.210340371976184f / 128.0f;      // ln(10000)/d
    const float* xb = x + ((size_t)b * CH * TT + t) * FB;
    for (int idx = tid; idx < FB * CH; idx += 512) {
        int f = idx & 127, c = idx >> 7;                 // contiguous in f => coalesced
        float v = xb[(size_t)c * (TT * FB) + f];
        float ang = (float)f * __expf(-(float)(c & ~1) * k10k);
        v += (c & 1) ? __cosf(ang) : __sinf(ang);
        Hs[f * HS + c] = (__bf16)v;
    }

    for (int l = 0; l < NLAYER; ++l) {
        wait_async_all();
        __syncthreads();   // B1: Wq/Wk/Wv staged (+input on l==0, +prev layer on l>0)

        // ---- Q/K/V projections: wave (w,s) does cols s*64..s*64+63 of band w ----
        for (int tc = 0; tc < 4; ++tc) {
            int n0 = (s * 4 + tc) << 4;
            f32x8 aq = {}, ak = {}, av = {};
#pragma unroll
            for (int kk = 0; kk < CH; kk += 32) {
                bf16x16 a = frag_a(Hs, HS, m0, kk, lane);
                aq = wmma_bf16(a, frag_b(WqL, HS, n0, kk, lane), aq);
                ak = wmma_bf16(a, frag_b(WkL, HS, n0, kk, lane), ak);
                av = wmma_bf16(a, frag_b(WvL, HS, n0, kk, lane), av);
            }
            int N  = n0 + ll;
            int Mb = m0 + hi8;
            float qb = bq[l * 128 + N], kb = bk[l * 128 + N], vb = bv[l * 128 + N];
#pragma unroll
            for (int r = 0; r < 8; ++r) {
                Qs[(Mb + r) * HS + N] = (__bf16)((aq[r] + qb) * 0.125f);  // scale folded in
                Ks[(Mb + r) * HS + N] = (__bf16)(ak[r] + kb);
                Vt[N * HS + (Mb + r)] = (__bf16)(av[r] + vb);             // transposed store
            }
        }
        __syncthreads();   // B2: K/Vt cross-band; also frees WqL/WkL/WvL regions

        // ---- attention: wave (w,s) handles head s for band w ----
        {
            const int ho = s * DHEAD;

            // S = (Q/sqrt(d)) * K^T : full row band kept in registers
            f32x8 sacc[8];
#pragma unroll
            for (int tc = 0; tc < 8; ++tc) {
                int n0 = tc << 4;
                f32x8 acc = {};
#pragma unroll
                for (int kk = 0; kk < DHEAD; kk += 32) {
                    bf16x16 a  = frag_a(Qs, HS, m0, ho + kk, lane);
                    bf16x16 bb = frag_b(Ks, HS, n0, ho + kk, lane);
                    acc = wmma_bf16(a, bb, acc);
                }
                sacc[tc] = acc;
            }
            __syncthreads();   // B3: Qs/Ks fully consumed -> Qs slot free for Wo

            // Wo copy overlaps softmax + O-GEMM (ASYNCcnt)
            stage_weight_async(wT + (size_t)(l * 4 + 3) * 16384, (unsigned)OFF_Q, tid);

            // masked softmax entirely in registers (butterfly over the 16-lane half)
#pragma unroll
            for (int r = 0; r < 8; ++r) {
                int m = m0 + hi8 + r;
                float sv[8];
                float mx = -3.0e38f;
#pragma unroll
                for (int tc = 0; tc < 8; ++tc) {
                    int c = (tc << 4) + ll;
                    int d = m - c; d = d < 0 ? -d : d;
                    bool at = (d < 64) ? ((MASK_LO >> d) & 1ULL)
                                       : ((MASK_HI >> (d - 64)) & 1ULL);
                    sv[tc] = at ? sacc[tc][r] : -3.0e38f;
                    mx = fmaxf(mx, sv[tc]);
                }
#pragma unroll
                for (int off = 8; off; off >>= 1)
                    mx = fmaxf(mx, __shfl_xor(mx, off, 32));
                float e[8];
                float sum = 0.f;
#pragma unroll
                for (int tc = 0; tc < 8; ++tc) {
                    e[tc] = __expf(sv[tc] - mx);        // masked -> exp(-huge) == 0
                    sum += e[tc];
                }
#pragma unroll
                for (int off = 8; off; off >>= 1)
                    sum += __shfl_xor(sum, off, 32);
                float inv = 1.0f / sum;
#pragma unroll
                for (int tc = 0; tc < 8; ++tc)
                    Pss[m * HS + (tc << 4) + ll] = (__bf16)(e[tc] * inv);
            }
            // no barrier: P band is produced and consumed by this wave

            // O_h = P * V_h  (Vt is column-major B; synced at B2)
            for (int tc = 0; tc < 4; ++tc) {
                int n0 = tc << 4;
                f32x8 acc = {};
#pragma unroll
                for (int kk = 0; kk < FB; kk += 32) {
                    bf16x16 a  = frag_a(Pss, HS, m0, kk, lane);
                    bf16x16 bb = frag_b(Vt, HS, ho + n0, kk, lane);
                    acc = wmma_bf16(a, bb, acc);
                }
                int N  = n0 + ll;
                int Mb = m0 + hi8;
#pragma unroll
                for (int r = 0; r < 8; ++r)
                    Os[(Mb + r) * HS + ho + N] = (__bf16)acc[r];
            }
        }
        wait_async_all();      // own Wo copy portion done
        __syncthreads();       // B4: Os complete (both head-waves) + WoL visible

        // ---- output projection: H' = O * Wo^T + bo ----
        for (int tc = 0; tc < 4; ++tc) {
            int n0 = (s * 4 + tc) << 4;
            f32x8 acc = {};
#pragma unroll
            for (int kk = 0; kk < CH; kk += 32) {
                bf16x16 a  = frag_a(Os, HS, m0, kk, lane);
                bf16x16 bb = frag_b(WoL, HS, n0, kk, lane);
                acc = wmma_bf16(a, bb, acc);
            }
            int N  = n0 + ll;
            int Mb = m0 + hi8;
            float ob = bo[l * 128 + N];
            if (l < NLAYER - 1) {
#pragma unroll
                for (int r = 0; r < 8; ++r)
                    Hs[(Mb + r) * HS + N] = (__bf16)(acc[r] + ob);
            } else {
                // out[b, c=N, t, f=Mb+r] ; 8 consecutive f per lane
                float* op = out + (((size_t)(b * CH + N) * TT + t) * FB) + Mb;
#pragma unroll
                for (int r = 0; r < 8; ++r) op[r] = acc[r] + ob;
            }
        }
        __syncthreads();       // B5: Os/P regions free -> stage next layer's weights

        if (l < NLAYER - 1) {
            stage_weight_async(wT + (size_t)((l + 1) * 4 + 0) * 16384, (unsigned)OFF_O, tid);
            stage_weight_async(wT + (size_t)((l + 1) * 4 + 1) * 16384, (unsigned)OFF_P, tid);
            stage_weight_async(wT + (size_t)((l + 1) * 4 + 2) * 16384, (unsigned)(OFF_P + 34816), tid);
        }
    }
}

extern "C" void kernel_launch(void* const* d_in, const int* in_sizes, int n_in,
                              void* d_out, int out_size, void* d_ws, size_t ws_size,
                              hipStream_t stream) {
    (void)in_sizes; (void)n_in; (void)out_size; (void)ws_size;
    const float* x  = (const float*)d_in[0];
    const float* Wq = (const float*)d_in[1];
    const float* bq = (const float*)d_in[2];
    const float* Wk = (const float*)d_in[3];
    const float* bk = (const float*)d_in[4];
    const float* Wv = (const float*)d_in[5];
    const float* bv = (const float*)d_in[6];
    const float* Wo = (const float*)d_in[7];
    const float* bo = (const float*)d_in[8];
    float*  out = (float*)d_out;
    __bf16* wT  = (__bf16*)d_ws;           // 16 * 128*128 bf16 = 512 KB

    (void)hipFuncSetAttribute((const void*)ha_attn,
                              hipFuncAttributeMaxDynamicSharedMemorySize, SMEM_BYTES);

    prep_weights<<<16, 256, 0, stream>>>(Wq, Wk, Wv, Wo, wT);
    ha_attn<<<4 * TT, 512, SMEM_BYTES, stream>>>(x, wT, bq, bk, bv, bo, out);
}
